// Einsum_71571335021246
// MI455X (gfx1250) — matrix operand order repaired and checked by reading, not compiled, not run on hardware
//
#include <hip/hip_runtime.h>

// ---------------------------------------------------------------------------
// Batched GEMM  C[b] = A[b] (2048x128) * B[b] (128x2048), fp32 in/out.
//
// bf16 "split-3" (hi/lo decomposition, drop lo*lo) on v_wmma_f32_16x16x32_bf16:
// near-fp32 accuracy at bf16 matrix-core rate.
//
// Fast path (67MB workspace): pre-split A/B once into bf16 hi/lo (B transposed
// to [N][K]), then a conversion-free WMMA GEMM with double-buffered
// GLOBAL_LOAD_ASYNC_TO_LDS_B128 staging (ASYNCcnt-gated). Fallback: fused
// split+GEMM with register staging.
// ---------------------------------------------------------------------------

typedef __bf16 bf16_t;
typedef bf16_t v16bf __attribute__((ext_vector_type(16)));
typedef float  v8f   __attribute__((ext_vector_type(8)));
typedef unsigned short ushort_t;

#define BATCH 32
#define DIM_I 2048
#define DIM_K 128
#define DIM_J 2048

// ---- CDNA5 async global->LDS copy (probe-guarded) -------------------------
#if defined(__has_builtin)
#if __has_builtin(__builtin_amdgcn_global_load_async_to_lds_b128) && \
    __has_builtin(__builtin_amdgcn_s_wait_asynccnt)
#define USE_ASYNC_COPY 1
#endif
#endif

#ifdef USE_ASYNC_COPY
// Builtin prototype (from clang diagnostic): param0 is
//   int __attribute__((vector_size(16))) AS(1) *   (global int4*)
typedef int v4i_vs __attribute__((vector_size(16)));
typedef __attribute__((address_space(1))) v4i_vs gv4i_t;
typedef __attribute__((address_space(3))) v4i_vs lv4i_t;
// Per-lane 16B copy global->LDS, tracked by ASYNCcnt (no VGPR round trip).
// LDS offset == low 32 bits of the flat pointer (LDS aperture: addr[31:0]).
__device__ __forceinline__ void async_cp_b128(const void* gsrc, void* ldst) {
  __builtin_amdgcn_global_load_async_to_lds_b128(
      (gv4i_t*)(unsigned long long)gsrc,
      (lv4i_t*)(unsigned int)(unsigned long long)ldst, 0, 0);
}
#define WAIT_ASYNC(n) __builtin_amdgcn_s_wait_asynccnt(n)
#else
#define WAIT_ASYNC(n) ((void)0)
#endif

union FragU {
  v16bf v;
  unsigned int u[8];
};

__device__ __forceinline__ ushort_t f32_to_bf16(float f) {
  unsigned int u = __float_as_uint(f);
  u += 0x7FFFu + ((u >> 16) & 1u);   // round-to-nearest-even
  return (ushort_t)(u >> 16);
}
__device__ __forceinline__ float bf16_to_f32(ushort_t h) {
  return __uint_as_float(((unsigned int)h) << 16);
}
__device__ __forceinline__ void split_bf16(float f, ushort_t& hi, ushort_t& lo) {
  hi = f32_to_bf16(f);
  lo = f32_to_bf16(f - bf16_to_f32(hi));
}
__device__ __forceinline__ void split_pack4(float x, float y, float z, float w,
                                            uint2& ph, uint2& pl) {
  ushort_t h0, h1, h2, h3, l0, l1, l2, l3;
  split_bf16(x, h0, l0); split_bf16(y, h1, l1);
  split_bf16(z, h2, l2); split_bf16(w, h3, l3);
  ph.x = (unsigned)h0 | ((unsigned)h1 << 16);
  ph.y = (unsigned)h2 | ((unsigned)h3 << 16);
  pl.x = (unsigned)l0 | ((unsigned)l1 << 16);
  pl.y = (unsigned)l2 | ((unsigned)l3 << 16);
}

// A-operand fragment (16-bit A 16x32 layout): lanes 0-15 hold K {0..7,16..23},
// lanes 16-31 hold K {8..15,24..31}; row = lane&15. p -> [row][k0 + koffA].
__device__ __forceinline__ void load_frag_a(const ushort_t* p, FragU& f) {
  uint4 x = *(const uint4*)p;         // K = k0+koffA .. +7
  uint4 y = *(const uint4*)(p + 16);  // K = k0+16+koffA .. +7
  f.u[0] = x.x; f.u[1] = x.y; f.u[2] = x.z; f.u[3] = x.w;
  f.u[4] = y.x; f.u[5] = y.y; f.u[6] = y.z; f.u[7] = y.w;
}
// B-operand fragment (16-bit B 32x16 layout, same K order as A): lanes 0-15 hold
// K {0..7,16..23}, lanes 16-31 hold K {8..15,24..31}. p -> [col][k0+koffB].
__device__ __forceinline__ void load_frag_b(const ushort_t* p, FragU& f) {
  uint4 x = *(const uint4*)p;         // K + 0..7
  uint4 y = *(const uint4*)(p + 16);  // K + 16..23
  f.u[0] = x.x; f.u[1] = x.y; f.u[2] = x.z; f.u[3] = x.w;
  f.u[4] = y.x; f.u[5] = y.y; f.u[6] = y.z; f.u[7] = y.w;
}

__device__ __forceinline__ v8f wmma3(FragU ah, FragU al, FragU bh, FragU bl, v8f c) {
  c = __builtin_amdgcn_wmma_f32_16x16x32_bf16(false, ah.v, false, bh.v, (short)0, c, false, false);
  c = __builtin_amdgcn_wmma_f32_16x16x32_bf16(false, ah.v, false, bl.v, (short)0, c, false, false);
  c = __builtin_amdgcn_wmma_f32_16x16x32_bf16(false, al.v, false, bh.v, (short)0, c, false, false);
  return c;
}

// ===========================================================================
// Pre-pass 1: split A (row-major [b][M][K]) into bf16 hi/lo, same layout.
// ===========================================================================
__global__ __launch_bounds__(256)
void split_a_kernel(const float* __restrict__ A,
                    ushort_t* __restrict__ Ah, ushort_t* __restrict__ Al) {
  size_t e4 = (size_t)blockIdx.x * 256 + threadIdx.x;  // float4 index
  const float4 v = ((const float4*)A)[e4];
  uint2 ph, pl;
  split_pack4(v.x, v.y, v.z, v.w, ph, pl);
  ((uint2*)Ah)[e4] = ph;
  ((uint2*)Al)[e4] = pl;
}

// ===========================================================================
// Pre-pass 2: split + transpose B ([b][K][J] f32) -> BT hi/lo ([b][J][K] bf16)
// via a 64x64 LDS tile (coalesced on both global sides).
// ===========================================================================
__global__ __launch_bounds__(256)
void split_bt_kernel(const float* __restrict__ B,
                     ushort_t* __restrict__ BTh, ushort_t* __restrict__ BTl) {
  __shared__ __align__(16) float tile[64][68];  // padded for bank spread
  const int tid = threadIdx.x;
  const int n0 = blockIdx.x * 64;  // along J
  const int k0 = blockIdx.y * 64;  // along K
  const int b  = blockIdx.z;

  const float* Bg = B + (size_t)b * DIM_K * DIM_J + (size_t)k0 * DIM_J + n0;
#pragma unroll
  for (int it = 0; it < 16; ++it) {
    int e  = it * 256 + tid;
    int kk = e >> 6;       // 0..63
    int nn = e & 63;       // lanes contiguous in n => coalesced 128B reads
    tile[nn][kk] = Bg[(size_t)kk * DIM_J + nn];
  }
  __syncthreads();

  ushort_t* outh = BTh + (size_t)b * DIM_J * DIM_K + (size_t)n0 * DIM_K + k0;
  ushort_t* outl = BTl + (size_t)b * DIM_J * DIM_K + (size_t)n0 * DIM_K + k0;
#pragma unroll
  for (int it = 0; it < 4; ++it) {
    int q  = it * 256 + tid;      // 0..1023 quads
    int nn = q >> 4;              // 0..63
    int kq = (q & 15) << 2;       // 0..60 step 4
    const float4 v = *(const float4*)&tile[nn][kq];
    uint2 ph, pl;
    split_pack4(v.x, v.y, v.z, v.w, ph, pl);
    *(uint2*)(outh + (size_t)nn * DIM_K + kq) = ph;
    *(uint2*)(outl + (size_t)nn * DIM_K + kq) = pl;
  }
}

// ===========================================================================
// Fast GEMM: reads pre-split bf16 arrays. 128x64 C-tile per block, K in 4
// slices of 32, double-buffered async global->LDS staging (ASYNCcnt-gated).
// Each of 8 waves computes 32x32 (2x2 subtiles): 48 WMMAs/wave.
// LDS: 2 buffers * (A hi/lo 128x40 + B hi/lo 64x40) * 2B = 60KB.
// ===========================================================================
#define LDKF 40   // padded K-slice stride (bf16 elems), 80B rows (16B aligned)

__global__ __launch_bounds__(256)
void bgemm_presplit(const ushort_t* __restrict__ Ahg, const ushort_t* __restrict__ Alg,
                    const ushort_t* __restrict__ Bhg, const ushort_t* __restrict__ Blg,
                    float* __restrict__ C) {
  __shared__ __align__(16) ushort_t AhL[2][128][LDKF];
  __shared__ __align__(16) ushort_t AlL[2][128][LDKF];
  __shared__ __align__(16) ushort_t BhL[2][64][LDKF];
  __shared__ __align__(16) ushort_t BlL[2][64][LDKF];

  const int tid  = threadIdx.x;
  const int lane = tid & 31;
  const int wave = tid >> 5;
  const int m_wave = (wave & 3) * 32;   // 4 waves along M
  const int n_wave = (wave >> 2) * 32;  // 2 waves along N

  const int batch = blockIdx.z;
  const int m0 = blockIdx.y * 128;
  const int n0 = blockIdx.x * 64;

  const size_t abase = (size_t)batch * DIM_I * DIM_K + (size_t)m0 * DIM_K;
  const size_t bbase = (size_t)batch * DIM_J * DIM_K + (size_t)n0 * DIM_K;
  const ushort_t* Ahb = Ahg + abase;
  const ushort_t* Alb = Alg + abase;
  const ushort_t* Bhb = Bhg + bbase;
  const ushort_t* Blb = Blg + bbase;
  float* Cbase = C + (size_t)batch * DIM_I * DIM_J;

  v8f acc[2][2];
#pragma unroll
  for (int i = 0; i < 2; ++i)
#pragma unroll
    for (int j = 0; j < 2; ++j) {
      v8f z = {0.f, 0.f, 0.f, 0.f, 0.f, 0.f, 0.f, 0.f};
      acc[i][j] = z;
    }

  const int koffA = (lane < 16) ? 0 : 8;
  const int koffB = (lane < 16) ? 0 : 8;
  const int mrow  = lane & 15;

  // Stage one K-slice (32 wide) of all four arrays into LDS buffer `buf`.
  // 6 x b128 per thread: 4 for A hi/lo (128 rows), 2 for B hi/lo (64 rows).
  auto stage = [&](int buf, int ks) {
#pragma unroll
    for (int it = 0; it < 2; ++it) {
      int e   = it * 256 + tid;     // 0..511
      int row = e >> 2;             // 0..127
      int c   = (e & 3) << 3;       // 0,8,16,24 bf16 elems
      size_t go = (size_t)row * DIM_K + (size_t)ks + c;
#ifdef USE_ASYNC_COPY
      async_cp_b128(Ahb + go, &AhL[buf][row][c]);
      async_cp_b128(Alb + go, &AlL[buf][row][c]);
#else
      *(uint4*)&AhL[buf][row][c] = *(const uint4*)(Ahb + go);
      *(uint4*)&AlL[buf][row][c] = *(const uint4*)(Alb + go);
#endif
    }
    {
      int row = tid >> 2;           // 0..63
      int c   = (tid & 3) << 3;
      size_t go = (size_t)row * DIM_K + (size_t)ks + c;
#ifdef USE_ASYNC_COPY
      async_cp_b128(Bhb + go, &BhL[buf][row][c]);
      async_cp_b128(Blb + go, &BlL[buf][row][c]);
#else
      *(uint4*)&BhL[buf][row][c] = *(const uint4*)(Bhb + go);
      *(uint4*)&BlL[buf][row][c] = *(const uint4*)(Blb + go);
#endif
    }
  };

  stage(0, 0);  // preamble prefetch

  for (int ks = 0; ks < 4; ++ks) {
    const int cur = ks & 1;
    if (ks < 3) {
      stage(cur ^ 1, (ks + 1) * 32);  // prefetch next slice into idle buffer
      // In-order ASYNCcnt completion: <=6 outstanding => slice `ks` landed.
      WAIT_ASYNC(6);
    } else {
      WAIT_ASYNC(0);
    }
    __syncthreads();  // all waves' copies visible

    FragU bh[2], bl[2];
#pragma unroll
    for (int nn = 0; nn < 2; ++nn) {
      load_frag_b(&BhL[cur][n_wave + nn * 16 + mrow][koffB], bh[nn]);
      load_frag_b(&BlL[cur][n_wave + nn * 16 + mrow][koffB], bl[nn]);
    }
#pragma unroll
    for (int mm = 0; mm < 2; ++mm) {
      FragU ah, al;
      load_frag_a(&AhL[cur][m_wave + mm * 16 + mrow][koffA], ah);
      load_frag_a(&AlL[cur][m_wave + mm * 16 + mrow][koffA], al);
#pragma unroll
      for (int nn = 0; nn < 2; ++nn)
        acc[mm][nn] = wmma3(ah, al, bh[nn], bl[nn], acc[mm][nn]);
    }
    __syncthreads();  // protect buffer `cur` before iter ks+1 prefetches into it
  }

  // writeback per 16x16 f32 C/D layout
#pragma unroll
  for (int mm = 0; mm < 2; ++mm) {
    int rbase = m0 + m_wave + mm * 16 + ((lane < 16) ? 0 : 8);
#pragma unroll
    for (int nn = 0; nn < 2; ++nn) {
      int col = n0 + n_wave + nn * 16 + mrow;
      float* cp = Cbase + (size_t)rbase * DIM_J + col;
#pragma unroll
      for (int i = 0; i < 8; ++i)
        cp[(size_t)i * DIM_J] = acc[mm][nn][i];
    }
  }
}

// ===========================================================================
// Fallback: fused split+GEMM (no workspace). 128x64 tile, K in 2 halves.
// ===========================================================================
#define TMF 128
#define TNF 64
#define KHF 64
#define LDK 72

__global__ __launch_bounds__(256)
void bgemm_fused(const float* __restrict__ A,
                 const float* __restrict__ B,
                 float* __restrict__ C) {
  __shared__ __align__(16) ushort_t Ah[TMF][LDK];
  __shared__ __align__(16) ushort_t Al[TMF][LDK];
  __shared__ __align__(16) ushort_t Bh[TNF][LDK];
  __shared__ __align__(16) ushort_t Bl[TNF][LDK];

  const int tid  = threadIdx.x;
  const int lane = tid & 31;
  const int wave = tid >> 5;
  const int m_wave = (wave & 3) * 32;
  const int n_wave = (wave >> 2) * 32;

  const int batch = blockIdx.z;
  const int m0 = blockIdx.y * TMF;
  const int n0 = blockIdx.x * TNF;

  const float* Abase = A + (size_t)batch * DIM_I * DIM_K + (size_t)m0 * DIM_K;
  const float* Bbase = B + (size_t)batch * DIM_K * DIM_J + n0;
  float*       Cbase = C + (size_t)batch * DIM_I * DIM_J;

  v8f acc[2][2];
#pragma unroll
  for (int i = 0; i < 2; ++i)
#pragma unroll
    for (int j = 0; j < 2; ++j) {
      v8f z = {0.f, 0.f, 0.f, 0.f, 0.f, 0.f, 0.f, 0.f};
      acc[i][j] = z;
    }

  const int koffA = (lane < 16) ? 0 : 8;
  const int koffB = (lane < 16) ? 0 : 8;
  const int mrow  = lane & 15;

  for (int khalf = 0; khalf < DIM_K; khalf += KHF) {
    __syncthreads();
    {
      const float* Ag = Abase + khalf;
#pragma unroll
      for (int it = 0; it < 8; ++it) {
        int idx = it * 256 + tid;
        int row = idx >> 4;
        int c4  = (idx & 15) << 2;
        const float4 vv = *(const float4*)(Ag + (size_t)row * DIM_K + c4);
        uint2 ph, pl;
        split_pack4(vv.x, vv.y, vv.z, vv.w, ph, pl);
        *(uint2*)&Ah[row][c4] = ph;
        *(uint2*)&Al[row][c4] = pl;
      }
    }
    {
      const float* Bg = Bbase + (size_t)khalf * DIM_J;
#pragma unroll
      for (int it = 0; it < 4; ++it) {
        int q  = it * 256 + tid;
        int n  = q & (TNF - 1);
        int kb = q >> 6;
        float f0 = Bg[(size_t)(4 * kb + 0) * DIM_J + n];
        float f1 = Bg[(size_t)(4 * kb + 1) * DIM_J + n];
        float f2 = Bg[(size_t)(4 * kb + 2) * DIM_J + n];
        float f3 = Bg[(size_t)(4 * kb + 3) * DIM_J + n];
        uint2 ph, pl;
        split_pack4(f0, f1, f2, f3, ph, pl);
        *(uint2*)&Bh[n][4 * kb] = ph;
        *(uint2*)&Bl[n][4 * kb] = pl;
      }
    }
    __syncthreads();

#pragma unroll
    for (int k0 = 0; k0 < KHF; k0 += 32) {
      FragU bh[2], bl[2];
#pragma unroll
      for (int nn = 0; nn < 2; ++nn) {
        load_frag_b(&Bh[n_wave + nn * 16 + mrow][k0 + koffB], bh[nn]);
        load_frag_b(&Bl[n_wave + nn * 16 + mrow][k0 + koffB], bl[nn]);
      }
#pragma unroll
      for (int mm = 0; mm < 2; ++mm) {
        FragU ah, al;
        load_frag_a(&Ah[m_wave + mm * 16 + mrow][k0 + koffA], ah);
        load_frag_a(&Al[m_wave + mm * 16 + mrow][k0 + koffA], al);
#pragma unroll
        for (int nn = 0; nn < 2; ++nn)
          acc[mm][nn] = wmma3(ah, al, bh[nn], bl[nn], acc[mm][nn]);
      }
    }
  }

#pragma unroll
  for (int mm = 0; mm < 2; ++mm) {
    int rbase = m0 + m_wave + mm * 16 + ((lane < 16) ? 0 : 8);
#pragma unroll
    for (int nn = 0; nn < 2; ++nn) {
      int col = n0 + n_wave + nn * 16 + mrow;
      float* cp = Cbase + (size_t)rbase * DIM_J + col;
#pragma unroll
      for (int i = 0; i < 8; ++i)
        cp[(size_t)i * DIM_J] = acc[mm][nn][i];
    }
  }
}

// ===========================================================================
extern "C" void kernel_launch(void* const* d_in, const int* in_sizes, int n_in,
                              void* d_out, int out_size, void* d_ws, size_t ws_size,
                              hipStream_t stream) {
  (void)in_sizes; (void)n_in; (void)out_size;
  const float* A = (const float*)d_in[0];
  const float* B = (const float*)d_in[1];
  float* C = (float*)d_out;

  const size_t elems = (size_t)BATCH * DIM_I * DIM_K;  // == B elem count too
  const size_t need  = elems * 2 /*bytes per bf16*/ * 4 /*Ah,Al,BTh,BTl*/;

  if (ws_size >= need && d_ws != nullptr) {
    ushort_t* Ah  = (ushort_t*)d_ws;
    ushort_t* Al  = Ah + elems;
    ushort_t* BTh = Al + elems;
    ushort_t* BTl = BTh + elems;
    // pre-split (once per element)
    split_a_kernel<<<dim3((unsigned)(elems / 1024)), dim3(256), 0, stream>>>(A, Ah, Al);
    split_bt_kernel<<<dim3(DIM_J / 64, DIM_K / 64, BATCH), dim3(256), 0, stream>>>(B, BTh, BTl);
    // conversion-free WMMA GEMM, double-buffered async staging
    bgemm_presplit<<<dim3(DIM_J / 64, DIM_I / 128, BATCH), dim3(256), 0, stream>>>(
        Ah, Al, BTh, BTl, C);
  } else {
    bgemm_fused<<<dim3(DIM_J / TNF, DIM_I / TMF, BATCH), dim3(256), 0, stream>>>(A, B, C);
  }
}
